// DFA_88158498718324
// MI455X (gfx1250) — matrix operand order repaired and checked by reading, not compiled, not run on hardware
//
#include <hip/hip_runtime.h>

// ---------------------------------------------------------------------------
// Submanifold sparse conv network on gfx1250 via v_wmma_f32_16x16x32_bf16.
// gather(nbr) -> bf16 LDS tile -> WMMA accumulate(fp32) -> fused epilogue.
// ---------------------------------------------------------------------------

typedef __attribute__((ext_vector_type(16))) __bf16 v16bf;
typedef __attribute__((ext_vector_type(8)))  float  v8f;

#define NVOX 262144
#define KTAPS 27

__device__ __forceinline__ unsigned short f2bf(float f) {
    unsigned u = __float_as_uint(f);
    unsigned r = (u + 0x7FFFu + ((u >> 16) & 1u)) >> 16;   // round-nearest-even
    return (unsigned short)r;
}

// ---------------------------------------------------------------------------
// Pack fp32 weights [T, CIN, COUT] into bf16 WMMA B-fragments:
// layout [t][kt][nt][lane][e] with K = kt*32 + (lane/16)*16 + e, N = nt*16 + lane%16
// so each lane's 16 bf16 values are 32 contiguous bytes.
// ---------------------------------------------------------------------------
__global__ void pack_wgt_bf16(const float* __restrict__ W,
                              unsigned short* __restrict__ dst,
                              int CIN, int COUT, int total) {
    int tid = blockIdx.x * blockDim.x + threadIdx.x;
    if (tid >= total) return;
    int e    = tid & 15;
    int lane = (tid >> 4) & 31;
    int coutTiles = COUT >> 4;
    int cinTiles  = CIN >> 5;
    int nt   = (tid >> 9) % coutTiles;
    int rest = (tid >> 9) / coutTiles;
    int kt   = rest % cinTiles;
    int t    = rest / cinTiles;
    int K = kt * 32 + (lane >> 4) * 16 + e;
    int Nc = nt * 16 + (lane & 15);
    dst[tid] = f2bf(W[(t * CIN + K) * COUT + Nc]);
}

// ---------------------------------------------------------------------------
// Generic gather-GEMM spconv tile kernel.
//   CIN, COUT in {32,64};  TAPS = 27 (with nbr) or 1 (dense 1x1)
//   RELU: relu epilogue; RES: += res[row*outStride+col]; IN2: gather (in+in2)
// Workgroup: 256 threads / 8 waves; each wave owns one 16x16 output tile.
// ---------------------------------------------------------------------------
template <int CIN, int COUT, int TAPS, bool USE_NBR, bool RELU, bool RES, bool IN2>
__global__ __launch_bounds__(256)
void spconv_wmma(const float* __restrict__ in,
                 const float* __restrict__ in2,
                 const int* __restrict__ nbr,
                 const unsigned short* __restrict__ pW,
                 float* __restrict__ out,
                 const float* __restrict__ res,
                 int outStride, int colOff) {
    constexpr int ROWS   = 2048 / COUT;        // voxels per workgroup
    constexpr int PITCH  = CIN + 8;            // bf16 elems; pitch*2 multiple of 16B
    constexpr int NT     = COUT / 16;
    constexpr int KSTEPS = CIN / 32;
    constexpr int VEC    = (ROWS * CIN) / 4;   // float4 gather chunks

    __shared__ unsigned short ldsA[ROWS * PITCH];

    const int tid  = threadIdx.x;
    const int lane = tid & 31;
    const int wave = tid >> 5;
    const int mt   = wave / NT;                // wave's M tile
    const int nt   = wave % NT;                // wave's N tile
    const int blockRow = blockIdx.x * ROWS;

    const int mrow = lane & 15;
    const int half = lane >> 4;

    v8f acc = {0.f, 0.f, 0.f, 0.f, 0.f, 0.f, 0.f, 0.f};

    for (int tap = 0; tap < TAPS; ++tap) {
        // ---- gather ROWS x CIN fp32 rows -> bf16 LDS tile -----------------
        for (int v = tid; v < VEC; v += 256) {
            int row = v / (CIN / 4);
            int c4  = v % (CIN / 4);
            int grow = blockRow + row;
            int idx  = USE_NBR ? nbr[grow * TAPS + tap] : grow;
            float4 g = ((const float4*)in)[idx * (CIN / 4) + c4];
            if (IN2) {
                float4 f2 = ((const float4*)in2)[idx * (CIN / 4) + c4];
                g.x += f2.x; g.y += f2.y; g.z += f2.z; g.w += f2.w;
            }
            unsigned p0 = ((unsigned)f2bf(g.y) << 16) | f2bf(g.x);
            unsigned p1 = ((unsigned)f2bf(g.w) << 16) | f2bf(g.z);
            uint2 pk; pk.x = p0; pk.y = p1;
            *(uint2*)&ldsA[row * PITCH + c4 * 4] = pk;     // 8B aligned
        }
        __syncthreads();

        // ---- WMMA over K -------------------------------------------------
#pragma unroll
        for (int kt = 0; kt < KSTEPS; ++kt) {
            // A: 16x32 bf16 tile, contiguous K order (lane half picks K sub-ranges)
            const unsigned short* aPtr =
                &ldsA[(mt * 16 + mrow) * PITCH + kt * 32 + half * 16];
            union { uint4 q[2]; v16bf v; } ua, ub;
            ua.q[0] = *(const uint4*)(aPtr);        // K = half*16 + 0..7
            ua.q[1] = *(const uint4*)(aPtr + 8);    // K = half*16 + 8..15
            // B: prepacked per-lane contiguous 32B
            const unsigned short* bPtr =
                pW + ((tap * KSTEPS + kt) * NT + nt) * 512 + lane * 16;
            ub.q[0] = *(const uint4*)(bPtr);
            ub.q[1] = *(const uint4*)(bPtr + 8);
            acc = __builtin_amdgcn_wmma_f32_16x16x32_bf16(
                false, ua.v, false, ub.v, (short)0, acc, false, false);
        }
        __syncthreads();
    }

    // ---- epilogue: 16x16 f32 C/D layout (VGPR v -> M = half*8 + v) -------
#pragma unroll
    for (int v = 0; v < 8; ++v) {
        int m   = half * 8 + v;
        int row = blockRow + mt * 16 + m;
        int col = colOff + nt * 16 + (lane & 15);
        float val = acc[v];
        if (RELU) val = fmaxf(val, 0.f);
        if (RES)  val += res[row * outStride + col];
        out[row * outStride + col] = val;
    }
}

// ---------------------------------------------------------------------------
// Host-side orchestration
// ---------------------------------------------------------------------------
static inline void pack_launch(const float* W, unsigned short* dst,
                               int T, int CIN, int COUT, hipStream_t s) {
    int total = T * CIN * COUT;
    pack_wgt_bf16<<<(total + 255) / 256, 256, 0, s>>>(W, dst, CIN, COUT, total);
}

extern "C" void kernel_launch(void* const* d_in, const int* in_sizes, int n_in,
                              void* d_out, int out_size, void* d_ws, size_t ws_size,
                              hipStream_t stream) {
    const float* feats = (const float*)d_in[0];   // [N,64]
    const int*   nbr   = (const int*)d_in[1];     // [N,27]
    const float* w1    = (const float*)d_in[2];   // [27,64,64]
    const float* w2    = (const float*)d_in[3];   // [27,64,64]
    const float* b00   = (const float*)d_in[4];   // [3,27,64,32]
    const float* b01   = (const float*)d_in[5];   // [3,27,32,32]
    const float* b10   = (const float*)d_in[6];   // [3,64,32]
    const float* b11   = (const float*)d_in[7];   // [3,27,32,32]
    const float* b12   = (const float*)d_in[8];   // [3,32,32]
    float* outp = (float*)d_out;

    // ---- workspace layout -------------------------------------------------
    char* ws = (char*)d_ws;
    float* xbuf = (float*)ws;                 ws += (size_t)NVOX * 64 * 4;  // x
    float* h0   = (float*)ws;                 ws += (size_t)NVOX * 64 * 4;  // ping
    float* h1   = outp;                       // reuse d_out as pong buffer
    float* tA   = (float*)ws;                 ws += (size_t)NVOX * 32 * 4;
    float* tB   = (float*)ws;                 ws += (size_t)NVOX * 32 * 4;
    unsigned short* pw1  = (unsigned short*)ws; ws += (size_t)27 * 64 * 64 * 2;
    unsigned short* pw2  = (unsigned short*)ws; ws += (size_t)27 * 64 * 64 * 2;
    unsigned short* pb00 = (unsigned short*)ws; ws += (size_t)81 * 64 * 32 * 2;
    unsigned short* pb01 = (unsigned short*)ws; ws += (size_t)81 * 32 * 32 * 2;
    unsigned short* pb11 = (unsigned short*)ws; ws += (size_t)81 * 32 * 32 * 2;
    unsigned short* pb10 = (unsigned short*)ws; ws += (size_t)3 * 64 * 32 * 2;
    unsigned short* pb12 = (unsigned short*)ws; ws += (size_t)3 * 32 * 32 * 2;
    (void)ws_size; (void)n_in; (void)in_sizes; (void)out_size;

    // ---- pack all weights to bf16 WMMA fragments --------------------------
    pack_launch(w1,  pw1,  27, 64, 64, stream);
    pack_launch(w2,  pw2,  27, 64, 64, stream);
    pack_launch(b00, pb00, 81, 64, 32, stream);
    pack_launch(b01, pb01, 81, 32, 32, stream);
    pack_launch(b11, pb11, 81, 32, 32, stream);
    pack_launch(b10, pb10,  3, 64, 32, stream);
    pack_launch(b12, pb12,  3, 32, 32, stream);

    const int G32 = NVOX / 32;   // ROWS=32 (Cout=64)
    const int G64 = NVOX / 64;   // ROWS=64 (Cout=32)

    // x = relu(conv1(feats))
    spconv_wmma<64, 64, 27, true, true, false, false>
        <<<G32, 256, 0, stream>>>(feats, nullptr, nbr, pw1, xbuf, nullptr, 64, 0);

    // 3x INR blocks: x -> h0 -> h1(d_out) -> h0
    const float* hin = xbuf;
    float* houts[3] = {h0, h1, h0};
    for (int i = 0; i < 3; ++i) {
        float* hout = houts[i];
        const unsigned short* W00 = pb00 + (size_t)i * 27 * 64 * 32;
        const unsigned short* W01 = pb01 + (size_t)i * 27 * 32 * 32;
        const unsigned short* W11 = pb11 + (size_t)i * 27 * 32 * 32;
        const unsigned short* W10 = pb10 + (size_t)i * 64 * 32;
        const unsigned short* W12 = pb12 + (size_t)i * 32 * 32;

        // branch 0: tA = relu(spconv64->32(hin)); hout[:, :32] = spconv32->32(tA) + hin[:, :32]
        spconv_wmma<64, 32, 27, true, true, false, false>
            <<<G64, 256, 0, stream>>>(hin, nullptr, nbr, W00, tA, nullptr, 32, 0);
        spconv_wmma<32, 32, 27, true, false, true, false>
            <<<G64, 256, 0, stream>>>(tA, nullptr, nbr, W01, hout, hin, 64, 0);

        // branch 1: tB = relu(hin @ w10); tA = relu(spconv32->32(tB));
        //           hout[:, 32:] = tA @ w12 + hin[:, 32:]
        spconv_wmma<64, 32, 1, false, true, false, false>
            <<<G64, 256, 0, stream>>>(hin, nullptr, nullptr, W10, tB, nullptr, 32, 0);
        spconv_wmma<32, 32, 27, true, true, false, false>
            <<<G64, 256, 0, stream>>>(tB, nullptr, nbr, W11, tA, nullptr, 32, 0);
        spconv_wmma<32, 32, 1, false, false, true, false>
            <<<G64, 256, 0, stream>>>(tA, nullptr, nullptr, W12, hout, hin, 64, 32);

        hin = hout;
    }

    // out = conv2(h_final + x)  (elementwise add fused into the gather)
    spconv_wmma<64, 64, 27, true, false, false, true>
        <<<G32, 256, 0, stream>>>(hin, xbuf, nbr, pw2, outp, nullptr, 64, 0);
}
